// FourierAttention_8512625181023
// MI455X (gfx1250) — matrix operand order repaired and checked by reading, not compiled, not run on hardware
//
#include <hip/hip_runtime.h>
#include <hip/hip_bf16.h>

// FourierAttention on MI455X (gfx1250), compile-only target.
//
// Reference = rfft -> gather 64 modes -> complex per-mode ExE mixing -> scatter
// -> irfft.  Only 64 of 2049 bins are live, so the FFTs collapse to partial-DFT
// GEMMs against precomputed cos/-sin bases, all run through
// V_WMMA_F32_16X16X32_F16 (f16 multiplicands, f32 accumulate).
//
// Roofline: ~36 GFLOP vs 512 MiB mandatory HBM traffic (read x once, write out
// once) => ~23 us at 23.3 TB/s; f16 WMMA compute is ~10 us, so the design goal
// is pure streaming.  fa_fwd (reads x) stages 32x128 f32 chunks into LDS with
// GLOBAL_LOAD_ASYNC_TO_LDS_B128 (double-buffered, s_wait_asynccnt) and
// software-pipelines the basis B-fragments in registers so no WMMA sits behind
// an s_wait_loadcnt 0.

constexpr int Bq  = 8;
constexpr int Hq  = 16;
constexpr int Lq  = 4096;
constexpr int Eq  = 128;
constexpr int Mq  = 64;
constexpr int BHq = Bq * Hq;   // 128

constexpr int CHUNK  = 32;     // l-rows staged per iteration
constexpr int ROWPAD = 132;    // f32 per LDS row: 16B-aligned rows, small pad

typedef __attribute__((ext_vector_type(16))) _Float16 v16h;
typedef __attribute__((ext_vector_type(8)))  _Float16 v8h;
typedef __attribute__((ext_vector_type(8)))  float    v8f;

__device__ __forceinline__ v16h pack16(v8h lo, v8h hi8) {
  v16h r;
#pragma unroll
  for (int j = 0; j < 8; ++j) { r[j] = lo[j]; r[8 + j] = hi8[j]; }
  return r;
}

__device__ __forceinline__ v8f wmma16(v16h a, v16h b, v8f c) {
  // (neg_a, A, neg_b, B, c_mod, C, reuse_a, reuse_b)
  return __builtin_amdgcn_wmma_f32_16x16x32_f16(false, a, false, b, (short)0, c,
                                                false, false);
}

// Async DMA: global (16B/lane) -> LDS, tracked by ASYNCcnt.
__device__ __forceinline__ void async_b128(unsigned ldsOff, const float* gaddr) {
  asm volatile("global_load_async_to_lds_b128 %0, %1, off"
               :: "v"(ldsOff), "v"(gaddr) : "memory");
}

// ---------------------------------------------------------------------------
// Kernel 1: build DFT basis tables (f16).
//   basMC[m][l] =  cos(2*pi*k_m*l/L)      (forward, B-fragment orientation)
//   basMS[m][l] = -sin(2*pi*k_m*l/L)
//   basLC[l][m] =  (c_m/L)*cos(...)       (inverse, A-fragment orientation)
//   basLS[l][m] = -(c_m/L)*sin(...)
// Phase computed exactly: (k*l) mod 4096 (L is a power of two).
// ---------------------------------------------------------------------------
__global__ __launch_bounds__(256) void fa_basis(const int* __restrict__ index,
                                                _Float16* __restrict__ basMC,
                                                _Float16* __restrict__ basMS,
                                                _Float16* __restrict__ basLC,
                                                _Float16* __restrict__ basLS) {
  int i = blockIdx.x * blockDim.x + threadIdx.x;
  if (i >= Lq * Mq) return;
  int l = i / Mq;
  int m = i - l * Mq;
  int k = index[m];
  int ph = (k * l) & (Lq - 1);
  float th = (6.283185307179586f / (float)Lq) * (float)ph;
  float s, c;
  __sincosf(th, &s, &c);
  basMC[(size_t)m * Lq + l] = (_Float16)c;
  basMS[(size_t)m * Lq + l] = (_Float16)(-s);
  float scale = ((k == 0) ? 1.0f : 2.0f) / (float)Lq;
  basLC[(size_t)l * Mq + m] = (_Float16)(scale * c);
  basLS[(size_t)l * Mq + m] = (_Float16)(-scale * s);
}

// ---------------------------------------------------------------------------
// Kernel 2: forward partial DFT.  One 512-thread block per (b,h):
//   wave = (e-tile 0..7) x (m-half 0..1), each wave owns 2 m-tiles x {re,im}.
//   Sel[e,m] = sum_l x[l,e] * basM{C,S}[m][l]    (K = L = 4096, 128 chunks)
// x chunks (32 l x 128 e, f32, 16 KiB) DMA'd to LDS (double-buffered);
// basis B-fragments register-double-buffered (prefetched one half-iteration
// ahead).  Chunk loop unrolled x2 so all pipe indices are compile-time.
// Output layout sel{R,I}[m][bh][e] (f16): mix GEMM A-fragments K-contiguous.
// ---------------------------------------------------------------------------
struct BFrags { v8h c[2][2]; v8h s[2][2]; };   // [m-tile][8-half chunk]

__device__ __forceinline__ void loadB(BFrags& f,
                                      const _Float16* __restrict__ basMC,
                                      const _Float16* __restrict__ basMS,
                                      int mhalf, int ln, int kB, int l0) {
#pragma unroll
  for (int t = 0; t < 2; ++t) {
    const int m = (mhalf * 2 + t) * 16 + ln;
    const _Float16* bc = basMC + (size_t)m * Lq + l0 + kB;
    const _Float16* bs = basMS + (size_t)m * Lq + l0 + kB;
    f.c[t][0] = *(const v8h*)bc;  f.c[t][1] = *(const v8h*)(bc + 16);
    f.s[t][0] = *(const v8h*)bs;  f.s[t][1] = *(const v8h*)(bs + 16);
  }
}

__device__ __forceinline__ v16h buildA(const float* __restrict__ tl,
                                       int kloA, int e) {
  v16h a;
#pragma unroll
  for (int j = 0; j < 8; ++j) {
    a[j]     = (_Float16)tl[(kloA + j) * ROWPAD + e];
    a[8 + j] = (_Float16)tl[(kloA + 16 + j) * ROWPAD + e];
  }
  return a;
}

__device__ __forceinline__ void doWMMA(v8f accR[2], v8f accI[2],
                                       v16h a, const BFrags& f) {
#pragma unroll
  for (int t = 0; t < 2; ++t) {
    accR[t] = wmma16(a, pack16(f.c[t][0], f.c[t][1]), accR[t]);
    accI[t] = wmma16(a, pack16(f.s[t][0], f.s[t][1]), accI[t]);
  }
}

__global__ __launch_bounds__(512) void fa_fwd(const float* __restrict__ x,
                                              const _Float16* __restrict__ basMC,
                                              const _Float16* __restrict__ basMS,
                                              _Float16* __restrict__ selR,
                                              _Float16* __restrict__ selI) {
  __shared__ float tile[2][CHUNK * ROWPAD];

  const int bh    = blockIdx.x;
  const int tid   = threadIdx.x;
  const int wave  = tid >> 5;
  const int etile = wave & 7;
  const int mhalf = wave >> 3;           // 0/1 -> m-tiles {0,1} / {2,3}
  const int lane  = tid & 31;
  const int ln    = lane & 15;
  const int hi    = lane >> 4;
  const int kloA  = hi ? 8 : 0;          // A-matrix 16-bit K interleave
  const int kB    = hi * 8;              // B-matrix 16-bit K interleave
  const int e     = etile * 16 + ln;

  const float* xb = x + (size_t)bh * Lq * Eq;
  const unsigned ldsBase  = (unsigned)(unsigned long long)(void*)&tile[0][0];
  const unsigned bufBytes = (unsigned)(CHUNK * ROWPAD * sizeof(float));

  // Per-thread DMA assignment: 1024 b128 transfers per chunk, 2 per thread.
  const int idx0 = tid, idx1 = tid + 512;
  const int l0r = idx0 >> 5, s0 = idx0 & 31;   // row, 16B segment
  const int l1r = idx1 >> 5, s1 = idx1 & 31;
  const unsigned ldsOff0 = (unsigned)((l0r * ROWPAD + s0 * 4) * 4);
  const unsigned ldsOff1 = (unsigned)((l1r * ROWPAD + s1 * 4) * 4);

  auto issueChunk = [&](int chunk, int buf) {
    const unsigned b = (unsigned)buf * bufBytes;
    const float* g = xb + (size_t)chunk * CHUNK * Eq;
    async_b128(ldsBase + b + ldsOff0, g + (size_t)l0r * Eq + s0 * 4);
    async_b128(ldsBase + b + ldsOff1, g + (size_t)l1r * Eq + s1 * 4);
  };

  v8f accR[2] = {{0.f}}, accI[2] = {{0.f}};
  BFrags bf0, bf1;

  // Prologue: basis fragments for chunk 0 + DMA chunk 0 -> buffer 0.
  loadB(bf0, basMC, basMS, mhalf, ln, kB, 0);
  issueChunk(0, 0);

  const int nChunks = Lq / CHUNK;        // 128 (even)
  for (int kk = 0; kk < nChunks; kk += 2) {
    // ---- half A: consume buf0 / bf0, prefetch chunk kk+1 ----
    issueChunk(kk + 1, 1);                               // kk+1 <= 127 always
    asm volatile("s_wait_asynccnt 0x2" ::: "memory");    // chunk kk landed
    __syncthreads();
    v16h a0 = buildA(&tile[0][0], kloA, e);
    loadB(bf1, basMC, basMS, mhalf, ln, kB, (kk + 1) * CHUNK);  // prefetch
    doWMMA(accR, accI, a0, bf0);
    __syncthreads();                                     // buf0 consumed

    // ---- half B: consume buf1 / bf1, prefetch chunk kk+2 ----
    if (kk + 2 < nChunks) {
      issueChunk(kk + 2, 0);
      asm volatile("s_wait_asynccnt 0x2" ::: "memory");  // chunk kk+1 landed
    } else {
      asm volatile("s_wait_asynccnt 0x0" ::: "memory");
    }
    __syncthreads();
    v16h a1 = buildA(&tile[1][0], kloA, e);
    if (kk + 2 < nChunks)
      loadB(bf0, basMC, basMS, mhalf, ln, kB, (kk + 2) * CHUNK);
    doWMMA(accR, accI, a1, bf1);
    __syncthreads();                                     // buf1 consumed
  }

  // C/D layout: VGPR r -> e_row = etile*16 + hi*8 + r, N = m.
#pragma unroll
  for (int t = 0; t < 2; ++t) {
    const int m = (mhalf * 2 + t) * 16 + ln;
    size_t off = ((size_t)m * BHq + bh) * Eq + etile * 16 + hi * 8;
    v8h hr, hx;
#pragma unroll
    for (int r = 0; r < 8; ++r) {
      hr[r] = (_Float16)accR[t][r];
      hx[r] = (_Float16)accI[t][r];
    }
    *(v8h*)(selR + off) = hr;
    *(v8h*)(selI + off) = hx;
  }
}

// ---------------------------------------------------------------------------
// Kernel 3: complex mode mixing, one block per (mode, bh-tile).
//   P[o] = sum_e Sel[e] * W[e,o]  (complex) == 4 real GEMMs, K = E = 128.
// Epilogue: Pr = RR - II, Pi = RI + IR (separate accumulators; subtract
// in the epilogue).  Output p{r,i}T[bh][o][m] so inverse B-frags are
// K(m)-contiguous.
// ---------------------------------------------------------------------------
__global__ __launch_bounds__(256) void fa_mix(const float* __restrict__ wr,
                                              const float* __restrict__ wi,
                                              const _Float16* __restrict__ selR,
                                              const _Float16* __restrict__ selI,
                                              _Float16* __restrict__ prT,
                                              _Float16* __restrict__ piT) {
  const int m    = blockIdx.x;
  const int bht  = blockIdx.y;
  const int wave = threadIdx.x >> 5;      // o-tile 0..7
  const int lane = threadIdx.x & 31;
  const int ln   = lane & 15;
  const int hi   = lane >> 4;
  const int kloA = hi ? 8 : 0;
  const int kB   = hi * 8;
  const int bhRow = bht * 16 + ln;
  const int o     = wave * 16 + ln;

  v8f aRR = {{0.f}}, aII = {{0.f}}, aRI = {{0.f}}, aIR = {{0.f}};

  for (int e0 = 0; e0 < Eq; e0 += 32) {
    const _Float16* sr = selR + ((size_t)m * BHq + bhRow) * Eq + e0 + kloA;
    const _Float16* si = selI + ((size_t)m * BHq + bhRow) * Eq + e0 + kloA;
    v16h aR = pack16(*(const v8h*)sr, *(const v8h*)(sr + 16));
    v16h aI = pack16(*(const v8h*)si, *(const v8h*)(si + 16));

    v16h bR, bI;
#pragma unroll
    for (int j = 0; j < 16; ++j) {
      size_t we = ((size_t)(e0 + kB + (j < 8 ? j : j + 8)) * Eq + o) * Mq + m;
      bR[j] = (_Float16)wr[we];
      bI[j] = (_Float16)wi[we];
    }
    aRR = wmma16(aR, bR, aRR);
    aII = wmma16(aI, bI, aII);
    aRI = wmma16(aR, bI, aRI);
    aIR = wmma16(aI, bR, aIR);
  }

#pragma unroll
  for (int r = 0; r < 8; ++r) {
    int bh = bht * 16 + hi * 8 + r;
    size_t off = ((size_t)bh * Eq + o) * Mq + m;
    prT[off] = (_Float16)(aRR[r] - aII[r]);
    piT[off] = (_Float16)(aRI[r] + aIR[r]);
  }
}

// ---------------------------------------------------------------------------
// Kernel 4: inverse partial DFT (the 256-MiB store kernel; sits on the HBM
// roofline).  One block per (l-tile, bh); wave w owns output channel tile w.
//   out[l,o] = sum_m basLC[l,m]*Pr[o,m] + basLS[l,m]*Pi[o,m]   (K = 64)
// ---------------------------------------------------------------------------
__global__ __launch_bounds__(256) void fa_inv(const _Float16* __restrict__ basLC,
                                              const _Float16* __restrict__ basLS,
                                              const _Float16* __restrict__ prT,
                                              const _Float16* __restrict__ piT,
                                              float* __restrict__ out) {
  const int lt   = blockIdx.x;
  const int bh   = blockIdx.y;
  const int wave = threadIdx.x >> 5;      // o-tile 0..7
  const int lane = threadIdx.x & 31;
  const int ln   = lane & 15;
  const int hi   = lane >> 4;
  const int kloA = hi ? 8 : 0;
  const int kB   = hi * 8;
  const int lRow = lt * 16 + ln;
  const int o    = wave * 16 + ln;

  v8f acc = {{0.f}};

#pragma unroll
  for (int m0 = 0; m0 < Mq; m0 += 32) {
    const _Float16* ac = basLC + (size_t)lRow * Mq + m0 + kloA;
    const _Float16* as = basLS + (size_t)lRow * Mq + m0 + kloA;
    v16h A1 = pack16(*(const v8h*)ac, *(const v8h*)(ac + 16));
    v16h A2 = pack16(*(const v8h*)as, *(const v8h*)(as + 16));

    const _Float16* bp = prT + ((size_t)bh * Eq + o) * Mq + m0 + kB;
    const _Float16* bq = piT + ((size_t)bh * Eq + o) * Mq + m0 + kB;
    v16h B1 = pack16(*(const v8h*)bp, *(const v8h*)(bp + 16));
    v16h B2 = pack16(*(const v8h*)bq, *(const v8h*)(bq + 16));

    acc = wmma16(A1, B1, acc);
    acc = wmma16(A2, B2, acc);
  }

  // Store: row l = lt*16 + hi*8 + r, col o; lanes 0-15 write 64B runs.
  float* op = out + ((size_t)bh * Lq + lt * 16 + hi * 8) * Eq + o;
#pragma unroll
  for (int r = 0; r < 8; ++r) op[(size_t)r * Eq] = acc[r];
}

// ---------------------------------------------------------------------------
// Launch: in_sizes order = {x, weights_real, weights_imag, index}.
// Workspace layout (bytes):
//   [0,512K)    basMC   f16 [M][L]
//   [512K,1M)   basMS
//   [1M,1.5M)   basLC   f16 [L][M] (c_m/L folded)
//   [1.5M,2M)   basLS
//   [2M,4M)     selR    f16 [M][BH][E]
//   [4M,6M)     selI
//   [6M,8M)     prT     f16 [BH][E][M]
//   [8M,10M)    piT
// ---------------------------------------------------------------------------
extern "C" void kernel_launch(void* const* d_in, const int* in_sizes, int n_in,
                              void* d_out, int out_size, void* d_ws, size_t ws_size,
                              hipStream_t stream) {
  const float* x   = (const float*)d_in[0];
  const float* wr  = (const float*)d_in[1];
  const float* wi  = (const float*)d_in[2];
  const int*   idx = (const int*)d_in[3];
  float* out = (float*)d_out;

  char* w = (char*)d_ws;
  _Float16* basMC = (_Float16*)(w + (size_t)0);
  _Float16* basMS = (_Float16*)(w + (size_t)512 * 1024);
  _Float16* basLC = (_Float16*)(w + (size_t)1024 * 1024);
  _Float16* basLS = (_Float16*)(w + (size_t)1536 * 1024);
  _Float16* selR  = (_Float16*)(w + (size_t)2048 * 1024);
  _Float16* selI  = (_Float16*)(w + (size_t)4096 * 1024);
  _Float16* prT   = (_Float16*)(w + (size_t)6144 * 1024);
  _Float16* piT   = (_Float16*)(w + (size_t)8192 * 1024);

  fa_basis<<<(Lq * Mq + 255) / 256, 256, 0, stream>>>(idx, basMC, basMS, basLC, basLS);
  fa_fwd<<<BHq, 512, 0, stream>>>(x, basMC, basMS, selR, selI);
  fa_mix<<<dim3(Mq, BHq / 16), 256, 0, stream>>>(wr, wi, selR, selI, prT, piT);
  fa_inv<<<dim3(Lq / 16, BHq), 256, 0, stream>>>(basLC, basLS, prT, piT, out);
}
